// GraphConvNet_32744830665554
// MI455X (gfx1250) — matrix operand order repaired and checked by reading, not compiled, not run on hardware
//
#include <hip/hip_runtime.h>

typedef __attribute__((ext_vector_type(16))) _Float16 v16h;
typedef __attribute__((ext_vector_type(8)))  _Float16 v8h;
typedef __attribute__((ext_vector_type(8)))  float    v8f;
typedef __attribute__((ext_vector_type(4)))  unsigned u32x4;
typedef __attribute__((ext_vector_type(8)))  unsigned u32x8;

#define NN 20000
#define NE 320000
#define LAT 128

// ---------------------------------------------------------------- WMMA core
__device__ __forceinline__ v8f wmma16(v16h a, v16h b, v8f c) {
    return __builtin_amdgcn_wmma_f32_16x16x32_f16(false, a, false, b, (short)0, c,
                                                  false, false);
}

// A-fragment (16x32 f16, ISA 7.12.2 layout) from a contiguous f16 feature row.
// lane<16: K = kbase+{0..7, 16..23};  lane>=16: K = kbase+{8..15, 24..31}
__device__ __forceinline__ v16h afrag_h(const _Float16* rp, int hi, int kbase) {
    const _Float16* p = rp + kbase + hi * 8;
    v16h a;
    *(v8h*)&a       = *(const v8h*)(p);
    *((v8h*)&a + 1) = *(const v8h*)(p + 16);
    return a;
}

__device__ __forceinline__ v16h afrag_f32(const float* rp, int hi, int kbase) {
    const float* p = rp + kbase + hi * 8;
    v16h a;
#pragma unroll
    for (int i = 0; i < 8; ++i) {
        a[i]     = (_Float16)p[i];
        a[8 + i] = (_Float16)p[16 + i];
    }
    return a;
}

// B-fragment (32x16 f16): lane<16 holds col N=lane, K=0..7,16..23; lane>=16 K=8..15,24..31.
__device__ __forceinline__ v16h bfrag(const _Float16* wf, int kt, int nt, int lane) {
    const _Float16* p = wf + ((size_t)(kt * 8 + nt)) * 512 + lane * 16;
    v16h b;
    *(v8h*)&b       = *(const v8h*)(p);
    *((v8h*)&b + 1) = *(const v8h*)(p + 8);
    return b;
}

__device__ __forceinline__ float gelu(float x) {
    float t = tanhf(0.7978845608028654f * (x + 0.044715f * x * x * x));
    return 0.5f * x * (1.0f + t);
}

__device__ __forceinline__ void zero_acc(v8f acc[8]) {
#pragma unroll
    for (int nt = 0; nt < 8; ++nt) acc[nt] = (v8f){0, 0, 0, 0, 0, 0, 0, 0};
}

// globals conditioning fragment: rows get g[0..3] at K 0..3 of this k-tile
__device__ __forceinline__ v16h gfrag(const float* g, int hi) {
    v16h a;
#pragma unroll
    for (int i = 0; i < 16; ++i) a[i] = (_Float16)0.0f;
#pragma unroll
    for (int i = 0; i < 4; ++i)
        a[i] = (_Float16)((hi == 0) ? g[i] : 0.0f);
    return a;
}

// bias + gelu, write 16x128 f16 tile to LDS row-major (A-fragment source)
__device__ __forceinline__ void bias_gelu_to_h(v8f acc[8], const float* bias,
                                               _Float16* hdst, int lane) {
    int cl = lane & 15, hi = lane >> 4;
#pragma unroll
    for (int nt = 0; nt < 8; ++nt) {
        int   col = nt * 16 + cl;
        float b   = bias[col];
#pragma unroll
        for (int v = 0; v < 8; ++v)
            hdst[(hi * 8 + v) * 128 + col] = (_Float16)gelu(acc[nt][v] + b);
    }
}

// hidden layer, single 16-row group: 16x128 (LDS f16) @ 128x128 -> acc
__device__ __forceinline__ void gemm_from_h(const _Float16* hsrc, const _Float16* wf,
                                            int lane, v8f acc[8]) {
    int row = lane & 15, hi = lane >> 4;
    const _Float16* rp = hsrc + row * 128;
#pragma unroll
    for (int kt = 0; kt < 4; ++kt) {
        v16h a = afrag_h(rp, hi, kt * 32);
#pragma unroll
        for (int nt = 0; nt < 8; ++nt)
            acc[nt] = wmma16(a, bfrag(wf, kt, nt, lane), acc[nt]);
    }
}

// hidden layer, dual 16-row groups sharing each B-fragment (x2 reuse)
__device__ __forceinline__ void gemm2_from_h(const _Float16* hbase, const _Float16* wf,
                                             int lane, v8f a0[8], v8f a1[8]) {
    int row = lane & 15, hi = lane >> 4;
    const _Float16* p0 = hbase + row * 128;
    const _Float16* p1 = hbase + (16 + row) * 128;
#pragma unroll
    for (int kt = 0; kt < 4; ++kt) {
        v16h x0 = afrag_h(p0, hi, kt * 32);
        v16h x1 = afrag_h(p1, hi, kt * 32);
#pragma unroll
        for (int nt = 0; nt < 8; ++nt) {
            v16h b = bfrag(wf, kt, nt, lane);
            a0[nt] = wmma16(x0, b, a0[nt]);
            a1[nt] = wmma16(x1, b, a1[nt]);
        }
    }
}

// layer-0 segment (4 k-tiles) for two gathered rows, shared B-fragments
__device__ __forceinline__ void l0_seg(const _Float16* r0, const _Float16* r1,
                                       const _Float16* wf0, int ktbase, int lane,
                                       v8f a0[8], v8f a1[8]) {
    int hi = lane >> 4;
#pragma unroll
    for (int t = 0; t < 4; ++t) {
        v16h x0 = afrag_h(r0, hi, t * 32);
        v16h x1 = afrag_h(r1, hi, t * 32);
#pragma unroll
        for (int nt = 0; nt < 8; ++nt) {
            v16h b = bfrag(wf0, ktbase + t, nt, lane);
            a0[nt] = wmma16(x0, b, a0[nt]);
            a1[nt] = wmma16(x1, b, a1[nt]);
        }
    }
}

// ------------------------------------------------- Tensor Data Mover staging
// 1-row 2D descriptor: ndw dwords, global -> LDS (ISA 08_async_tensor 8.3/8.4)
__device__ __forceinline__ void tdm_load_to_lds(unsigned lds_addr, const void* gsrc,
                                                unsigned ndw) {
    unsigned long long ga = (unsigned long long)gsrc;
    u32x4 g0;
    g0[0] = 1u;                                   // count=1, user mode
    g0[1] = lds_addr;                             // lds_addr
    g0[2] = (unsigned)ga;                         // global_addr[31:0]
    g0[3] = (unsigned)((ga >> 32) & 0x1FFFFFFu) | (2u << 30);  // addr hi | type=2
    u32x8 g1;
    g1[0] = 0x20000u;                             // data_size = 4B, no multicast
    g1[1] = (ndw & 0xFFFFu) << 16;                // tensor_dim0 lo16
    g1[2] = ((ndw >> 16) & 0xFFFFu) | (1u << 16); // tensor_dim0 hi | tensor_dim1=1
    g1[3] = (ndw & 0xFFFFu) << 16;                // tile_dim0 = ndw
    g1[4] = 1u;                                   // tile_dim1 = 1
    g1[5] = ndw;                                  // tensor_dim0_stride lo32
    g1[6] = 0u;
    g1[7] = 0u;
    asm volatile("tensor_load_to_lds %0, %1" : : "s"(g0), "s"(g1) : "memory");
}

// --------------------------------------------------------------- weight pack
__global__ void pack_weight_kernel(const float* __restrict__ W, _Float16* __restrict__ frag,
                                   int din, int ktiles) {
    int i = blockIdx.x * blockDim.x + threadIdx.x;
    int total = ktiles * 4096;
    if (i >= total) return;
    int kt   = i >> 12;
    int rem  = i & 4095;
    int nt   = rem >> 9;
    int lane = (rem >> 4) & 31;
    int h    = i & 15;
    int m = lane & 15, hi = lane >> 4;
    int k = kt * 32 + ((h < 8) ? (hi * 8 + h) : (16 + hi * 8 + (h - 8)));
    float v = (k < din) ? W[(size_t)k * 128 + nt * 16 + m] : 0.0f;
    frag[i] = (_Float16)v;
}

__global__ void zero_kernel(float* __restrict__ p, int n) {
    int i = blockIdx.x * blockDim.x + threadIdx.x;
    if (i < n) p[i] = 0.0f;
}

// ---------------------------------------------------------------- embed
__global__ void embed_kernel(const float* __restrict__ x, const float* __restrict__ W,
                             const float* __restrict__ b, float* __restrict__ nf,
                             _Float16* __restrict__ nh) {
    int i = blockIdx.x * blockDim.x + threadIdx.x;
    if (i >= NN * LAT) return;
    int n = i >> 7, c = i & 127;
    float v = b[c] + x[n * 3 + 0] * W[c] + x[n * 3 + 1] * W[128 + c] +
              x[n * 3 + 2] * W[256 + c];
    nf[i] = v;
    nh[i] = (_Float16)v;
}

// ---------------------------------------------------------------- edge MLP
// 4 waves/block; each wave owns a 32-edge tile (two 16-row A groups).
// Hidden-layer weights staged to LDS by TDM once per block.
__global__ __launch_bounds__(128) void edge_mlp_kernel(
    const int* __restrict__ senders, const int* __restrict__ receivers,
    const float* __restrict__ g, const _Float16* __restrict__ nodes_h,
    float* __restrict__ edges_acc, _Float16* __restrict__ edges_h,
    float* __restrict__ recv_agg,
    const _Float16* __restrict__ wf0, const float* __restrict__ b0,
    const _Float16* __restrict__ wf12,  // wf1|wf2 contiguous, 64KB
    const float* __restrict__ b1, const float* __restrict__ b2, int has_edges) {
    extern __shared__ char smem[];
    _Float16* wlds = (_Float16*)smem;  // 64KB: wf1 then wf2
    int wave = threadIdx.x >> 5, lane = threadIdx.x & 31;
    _Float16* h = (_Float16*)(smem + 65536) + wave * (32 * 128);  // 8KB/wave

    if (wave == 0) {
        tdm_load_to_lds((unsigned)(size_t)wlds, wf12, 16384);  // 64KB in dwords
        __builtin_amdgcn_s_wait_tensorcnt(0);
    }
    __syncthreads();
    const _Float16* wl1 = wlds;
    const _Float16* wl2 = wlds + 16384;

    int e0 = (blockIdx.x * 4 + wave) * 32;
    int row = lane & 15, hi = lane >> 4;

    const _Float16* sr0 = nodes_h + (size_t)senders[e0 + row] * 128;
    const _Float16* sr1 = nodes_h + (size_t)senders[e0 + 16 + row] * 128;
    const _Float16* rr0 = nodes_h + (size_t)receivers[e0 + row] * 128;
    const _Float16* rr1 = nodes_h + (size_t)receivers[e0 + 16 + row] * 128;
    const _Float16* er0 = edges_h + (size_t)(e0 + row) * 128;
    const _Float16* er1 = edges_h + (size_t)(e0 + 16 + row) * 128;

    // -------- layer 0: concat [edges?, sent, recv, g_pad]
    v8f acc0[8], acc1[8];
    zero_acc(acc0);
    zero_acc(acc1);
    int kt = 0;
    if (has_edges) {
        l0_seg(er0, er1, wf0, kt, lane, acc0, acc1);
        kt += 4;
    }
    l0_seg(sr0, sr1, wf0, kt, lane, acc0, acc1);
    kt += 4;
    l0_seg(rr0, rr1, wf0, kt, lane, acc0, acc1);
    kt += 4;
    {
        v16h a = gfrag(g, hi);
#pragma unroll
        for (int nt = 0; nt < 8; ++nt) {
            v16h b = bfrag(wf0, kt, nt, lane);
            acc0[nt] = wmma16(a, b, acc0[nt]);
            acc1[nt] = wmma16(a, b, acc1[nt]);
        }
    }
    bias_gelu_to_h(acc0, b0, h, lane);
    bias_gelu_to_h(acc1, b0, h + 16 * 128, lane);

    // -------- layer 1 (B from LDS)
    zero_acc(acc0);
    zero_acc(acc1);
    gemm2_from_h(h, wl1, lane, acc0, acc1);
    bias_gelu_to_h(acc0, b1, h, lane);
    bias_gelu_to_h(acc1, b1, h + 16 * 128, lane);

    // -------- layer 2 (B from LDS) + edge accumulate + scatter-add
    zero_acc(acc0);
    zero_acc(acc1);
    gemm2_from_h(h, wl2, lane, acc0, acc1);

    int rva[8], rvb[8];
#pragma unroll
    for (int v = 0; v < 8; ++v) {
        rva[v] = receivers[e0 + hi * 8 + v];
        rvb[v] = receivers[e0 + 16 + hi * 8 + v];
    }
#pragma unroll
    for (int nt = 0; nt < 8; ++nt) {
        int   col = nt * 16 + row;
        float bb  = b2[col];
#pragma unroll
        for (int v = 0; v < 8; ++v) {
            int m = hi * 8 + v;
            {
                float  x  = acc0[nt][v] + bb;
                size_t eo = (size_t)(e0 + m) * 128 + col;
                float  av = has_edges ? (edges_acc[eo] + x) : x;
                edges_acc[eo] = av;
                edges_h[eo]   = (_Float16)av;
                atomicAdd(&recv_agg[(size_t)rva[v] * 128 + col], x);
            }
            {
                float  x  = acc1[nt][v] + bb;
                size_t eo = (size_t)(e0 + 16 + m) * 128 + col;
                float  av = has_edges ? (edges_acc[eo] + x) : x;
                edges_acc[eo] = av;
                edges_h[eo]   = (_Float16)av;
                atomicAdd(&recv_agg[(size_t)rvb[v] * 128 + col], x);
            }
        }
    }
}

// ---------------------------------------------------------------- node MLP
__global__ __launch_bounds__(128) void node_mlp_kernel(
    const float* __restrict__ g, const float* __restrict__ recv_agg,
    float* __restrict__ nodes_f, _Float16* __restrict__ nodes_h,
    const _Float16* __restrict__ wf0, const float* __restrict__ b0,
    const _Float16* __restrict__ wf1, const float* __restrict__ b1,
    const _Float16* __restrict__ wf2, const float* __restrict__ b2,
    const float* __restrict__ lns, const float* __restrict__ lnb) {
    __shared__ _Float16 hbuf[4][16 * 128];
    __shared__ float    fbuf[4][16 * 128];
    int wave = threadIdx.x >> 5, lane = threadIdx.x & 31;
    int tile = blockIdx.x * 4 + wave;
    if (tile * 16 >= NN) return;
    int n0 = tile * 16;
    _Float16* h  = hbuf[wave];
    float*    fb = fbuf[wave];
    int row = lane & 15, hi = lane >> 4;
    const _Float16* nrow = nodes_h + (size_t)(n0 + row) * 128;
    const float*    arow = recv_agg + (size_t)(n0 + row) * 128;

    v8f acc[8];
    zero_acc(acc);
#pragma unroll
    for (int t = 0; t < 4; ++t) {
        v16h a = afrag_h(nrow, hi, t * 32);
#pragma unroll
        for (int nt = 0; nt < 8; ++nt)
            acc[nt] = wmma16(a, bfrag(wf0, t, nt, lane), acc[nt]);
    }
#pragma unroll
    for (int t = 0; t < 4; ++t) {
        v16h a = afrag_f32(arow, hi, t * 32);
#pragma unroll
        for (int nt = 0; nt < 8; ++nt)
            acc[nt] = wmma16(a, bfrag(wf0, 4 + t, nt, lane), acc[nt]);
    }
    {
        v16h a = gfrag(g, hi);
#pragma unroll
        for (int nt = 0; nt < 8; ++nt)
            acc[nt] = wmma16(a, bfrag(wf0, 8, nt, lane), acc[nt]);
    }
    bias_gelu_to_h(acc, b0, h, lane);

    zero_acc(acc);
    gemm_from_h(h, wf1, lane, acc);
    bias_gelu_to_h(acc, b1, h, lane);

    zero_acc(acc);
    gemm_from_h(h, wf2, lane, acc);

#pragma unroll
    for (int nt = 0; nt < 8; ++nt) {
        int   col = nt * 16 + row;
        float bb  = b2[col];
#pragma unroll
        for (int v = 0; v < 8; ++v) {
            int m = hi * 8 + v;
            fb[m * 128 + col] = nodes_f[(size_t)(n0 + m) * 128 + col] + acc[nt][v] + bb;
        }
    }

    if (lane < 16) {
        const float* r = fb + lane * 128;
        float mu = 0.0f;
#pragma unroll 4
        for (int c = 0; c < 128; ++c) mu += r[c];
        mu *= (1.0f / 128.0f);
        float var = 0.0f;
#pragma unroll 4
        for (int c = 0; c < 128; ++c) {
            float d = r[c] - mu;
            var += d * d;
        }
        var *= (1.0f / 128.0f);
        float  rs   = rsqrtf(var + 1e-6f);
        size_t base = (size_t)(n0 + lane) * 128;
#pragma unroll 4
        for (int c = 0; c < 128; ++c) {
            float y = (r[c] - mu) * rs * lns[c] + lnb[c];
            nodes_f[base + c] = y;
            nodes_h[base + c] = (_Float16)y;
        }
    }
}

// ---------------------------------------------------------------- decode
__global__ void decode_kernel(const float* __restrict__ nf, const float* __restrict__ W,
                              const float* __restrict__ b, float* __restrict__ out) {
    int n = blockIdx.x * blockDim.x + threadIdx.x;
    if (n >= NN) return;
    float s0 = b[0], s1 = b[1], s2 = b[2];
    const float* r = nf + (size_t)n * 128;
#pragma unroll 4
    for (int k = 0; k < 128; ++k) {
        float f = r[k];
        s0 += f * W[k * 3 + 0];
        s1 += f * W[k * 3 + 1];
        s2 += f * W[k * 3 + 2];
    }
    out[n * 3 + 0] = s0;
    out[n * 3 + 1] = s1;
    out[n * 3 + 2] = s2;
}

// ---------------------------------------------------------------- host
extern "C" void kernel_launch(void* const* d_in, const int* in_sizes, int n_in,
                              void* d_out, int out_size, void* d_ws, size_t ws_size,
                              hipStream_t stream) {
    (void)in_sizes; (void)n_in; (void)out_size; (void)ws_size;
    char*  ws  = (char*)d_ws;
    size_t off = 0;
    auto alloc = [&](size_t bytes) {
        char* p = ws + off;
        off = (off + bytes + 255) & ~(size_t)255;
        return p;
    };

    float*    nodes_f   = (float*)alloc((size_t)NN * LAT * 4);
    _Float16* nodes_h   = (_Float16*)alloc((size_t)NN * LAT * 2);
    float*    edges_acc = (float*)alloc((size_t)NE * LAT * 4);
    _Float16* edges_h   = (_Float16*)alloc((size_t)NE * LAT * 2);
    float*    recv_agg  = (float*)alloc((size_t)NN * LAT * 4);

    _Float16* efrag[4][3];
    _Float16* nfrag[4][3];
    for (int s = 0; s < 4; ++s) {
        int ek0 = (s == 0) ? 9 : 13;
        efrag[s][0] = (_Float16*)alloc((size_t)ek0 * 4096 * 2);
        efrag[s][1] = (_Float16*)alloc((size_t)4 * 4096 * 2);  // contiguous with
        efrag[s][2] = (_Float16*)alloc((size_t)4 * 4096 * 2);  // efrag[s][1] (32KB ea)
        nfrag[s][0] = (_Float16*)alloc((size_t)9 * 4096 * 2);
        nfrag[s][1] = (_Float16*)alloc((size_t)4 * 4096 * 2);
        nfrag[s][2] = (_Float16*)alloc((size_t)4 * 4096 * 2);
    }

    auto pack = [&](const void* W, _Float16* frag, int din) {
        int ktiles = (din + 31) / 32;
        int total  = ktiles * 4096;
        pack_weight_kernel<<<(total + 255) / 256, 256, 0, stream>>>(
            (const float*)W, frag, din, ktiles);
    };
    // inputs: 0:x 1:senders 2:receivers 3:globals 4:embedW 5:embedB
    // 6..29: edge_mlps, 30..53: node_mlps, 54..57: ln_scale, 58..61: ln_bias,
    // 62: decW, 63: decB
    for (int s = 0; s < 4; ++s) {
        int eb = 6 + s * 6, nb = 30 + s * 6;
        pack(d_in[eb + 0], efrag[s][0], s == 0 ? 260 : 388);
        pack(d_in[eb + 2], efrag[s][1], 128);
        pack(d_in[eb + 4], efrag[s][2], 128);
        pack(d_in[nb + 0], nfrag[s][0], 260);
        pack(d_in[nb + 2], nfrag[s][1], 128);
        pack(d_in[nb + 4], nfrag[s][2], 128);
    }

    embed_kernel<<<(NN * LAT + 255) / 256, 256, 0, stream>>>(
        (const float*)d_in[0], (const float*)d_in[4], (const float*)d_in[5],
        nodes_f, nodes_h);

    const int*   senders   = (const int*)d_in[1];
    const int*   receivers = (const int*)d_in[2];
    const float* g         = (const float*)d_in[3];

    for (int s = 0; s < 4; ++s) {
        int eb = 6 + s * 6, nb = 30 + s * 6;
        zero_kernel<<<(NN * LAT + 255) / 256, 256, 0, stream>>>(recv_agg, NN * LAT);
        // 32 edges/wave * 4 waves = 128 edges/block; NE/128 = 2500 blocks
        edge_mlp_kernel<<<NE / 128, 128, 96 * 1024, stream>>>(
            senders, receivers, g, nodes_h, edges_acc, edges_h, recv_agg,
            efrag[s][0], (const float*)d_in[eb + 1],
            efrag[s][1],  // wf1|wf2 contiguous
            (const float*)d_in[eb + 3], (const float*)d_in[eb + 5], s);
        node_mlp_kernel<<<(NN / 16 + 3) / 4, 128, 0, stream>>>(
            g, recv_agg, nodes_f, nodes_h,
            nfrag[s][0], (const float*)d_in[nb + 1],
            nfrag[s][1], (const float*)d_in[nb + 3],
            nfrag[s][2], (const float*)d_in[nb + 5],
            (const float*)d_in[54 + s], (const float*)d_in[58 + s]);
    }

    decode_kernel<<<(NN + 255) / 256, 256, 0, stream>>>(
        nodes_f, (const float*)d_in[62], (const float*)d_in[63], (float*)d_out);
}
